// LRGATConv_11811160064039
// MI455X (gfx1250) — matrix operand order repaired and checked by reading, not compiled, not run on hardware
//
#include <hip/hip_runtime.h>
#include <hip/hip_bf16.h>

// ---------------------------------------------------------------------------
// Problem constants (from reference)
// ---------------------------------------------------------------------------
#define N_GRAPHS  64
#define NODES_PER 64
#define EDGES_PER 128
#define N_NODES   4096   // N_GRAPHS * NODES_PER
#define N_EDGES   8192   // N_GRAPHS * EDGES_PER
#define F_N       64
#define F_E       32
#define H_N       128
#define H_E       128

typedef __attribute__((ext_vector_type(16))) __bf16 bf16x16;
typedef __attribute__((ext_vector_type(8)))  __bf16 bf16x8;
typedef __attribute__((ext_vector_type(8)))  float  f32x8;

// fp32 -> bf16, round to nearest even
static __device__ inline __bf16 f2bf(float f) {
    unsigned u = __float_as_uint(f);
    unsigned r = u + 0x7FFFu + ((u >> 16) & 1u);
    unsigned short s = (unsigned short)(r >> 16);
    return __builtin_bit_cast(__bf16, s);
}

// Async global->LDS copy, 16B per lane (tracked by ASYNCcnt).
static __device__ inline void async_copy_b128(unsigned lds_off, unsigned long long gaddr) {
    asm volatile("global_load_async_to_lds_b128 %0, %1, off"
                 :: "v"(lds_off), "v"(gaddr) : "memory");
}
static __device__ inline void wait_asynccnt0() {
    asm volatile("s_wait_asynccnt 0x0" ::: "memory");
}

// ---------------------------------------------------------------------------
// Prep: one-pass fp32 -> bf16 conversion + weight transpose.
//   xb  [M x K] bf16 row-major copies of x / edge_attr
//   WT  [N x K] bf16 transposed weights (so B fragments are contiguous in K)
// ---------------------------------------------------------------------------
__global__ void prep_bf16(const float* __restrict__ x,  const float* __restrict__ ea,
                          const float* __restrict__ Wh, const float* __restrict__ We,
                          __bf16* __restrict__ xb,  __bf16* __restrict__ eab,
                          __bf16* __restrict__ WhT, __bf16* __restrict__ WeT) {
    const int tid    = blockIdx.x * blockDim.x + threadIdx.x;
    const int stride = gridDim.x * blockDim.x;
    for (int i = tid; i < N_NODES * F_N; i += stride) xb[i]  = f2bf(x[i]);
    for (int i = tid; i < N_EDGES * F_E; i += stride) eab[i] = f2bf(ea[i]);
    for (int i = tid; i < F_N * H_N; i += stride) {          // Wh[k][n] -> WhT[n][k]
        int k = i / H_N, n = i % H_N;
        WhT[n * F_N + k] = f2bf(Wh[i]);
    }
    for (int i = tid; i < F_E * H_E; i += stride) {          // We[k][n] -> WeT[n][k]
        int k = i / H_E, n = i % H_E;
        WeT[n * F_E + k] = f2bf(We[i]);
    }
}

// ---------------------------------------------------------------------------
// C = relu(A[MxK]bf16 @ B(T)[NxK]bf16 + bias[N]); K templated -> unrolled WMMA.
// One wave per 16x16 tile. Fragment layouts (A/B use this code's contiguous K order):
//   A: lane<16 -> M=lane, K = kk+0..15;         lane>=16 -> K = kk+16..31
//   B: lane<16 -> N=lane, K = kk+0..15;         lane>=16 -> K = kk+16..31
//   D: VGPR r -> (M = m0+r (+8 hi half), N = n0 + (lane&15))
// Optional CfT_b: bf16 transposed copy of relu output ([N x M], contiguous-row
// stores straight from the D fragment: 8 consecutive rows per lane = 16B).
// ---------------------------------------------------------------------------
template <int K>
__global__ void gemm_relu_wmma(const __bf16* __restrict__ A,
                               const __bf16* __restrict__ BT,
                               const float*  __restrict__ bias,
                               float*  __restrict__ Cf,
                               __bf16* __restrict__ CfT_b,
                               int M, int N) {
    const int lane   = threadIdx.x & 31;
    const int warp   = threadIdx.x >> 5;
    const int tilesN = N >> 4;
    const int tile   = blockIdx.x * (blockDim.x >> 5) + warp;
    if (tile >= (M >> 4) * tilesN) return;
    const int m0   = (tile / tilesN) << 4;
    const int n0   = (tile % tilesN) << 4;
    const int half = lane >> 4;
    const int l15  = lane & 15;

    f32x8 acc = {};
#pragma unroll
    for (int kk = 0; kk < K; kk += 32) {
        const int abase = kk + (half ? 16 : 0);
        bf16x8 alo = *(const bf16x8*)(A + (m0 + l15) * K + abase);
        bf16x8 ahi = *(const bf16x8*)(A + (m0 + l15) * K + abase + 8);
        bf16x16 a  = __builtin_shufflevector(alo, ahi, 0,1,2,3,4,5,6,7,
                                             8,9,10,11,12,13,14,15);
        const int bbase = kk + (half ? 16 : 0);
        bf16x16 b = *(const bf16x16*)(BT + (n0 + l15) * K + bbase);
        acc = __builtin_amdgcn_wmma_f32_16x16x32_bf16(false, a, false, b,
                                                      (short)0, acc, false, false);
    }

    const int col = n0 + l15;
    const float bv = bias[col];
    bf16x8 tb;
#pragma unroll
    for (int r = 0; r < 8; ++r) {
        const int row = m0 + r + (half ? 8 : 0);
        float v = acc[r] + bv;
        v = v > 0.f ? v : 0.f;
        Cf[row * N + col] = v;
        tb[r] = f2bf(v);
    }
    if (CfT_b)   // rows m0+base..m0+base+7 are contiguous in the transposed layout
        *(bf16x8*)(CfT_b + (size_t)col * M + m0 + (half ? 8 : 0)) = tb;
}

// ---------------------------------------------------------------------------
// Node GAT update, one block per graph. h staged into LDS via ASYNC loads.
// score = relu(h[src].Wa[0:128] + h[dst].Wa[128:256] + ba)   (leaky(relu)=relu)
// segment softmax over src (self loops appended), out = sum coeff*h[dst].
// Deterministic: sequential per-node scans, no atomics.
// ---------------------------------------------------------------------------
__global__ void node_update(const float* __restrict__ h,
                            const int*   __restrict__ edge_index,
                            const float* __restrict__ Wa,
                            const float* __restrict__ ba,
                            float* __restrict__ nodes_out) {
    __shared__ float h_s[NODES_PER * H_N];                // 32 KB
    __shared__ int   src_s[EDGES_PER + NODES_PER];
    __shared__ int   dst_s[EDGES_PER + NODES_PER];
    __shared__ float score_s[EDGES_PER + NODES_PER];
    __shared__ float coeff_s[EDGES_PER + NODES_PER];
    __shared__ float dA[NODES_PER], dB[NODES_PER];
    __shared__ float mx[NODES_PER], den[NODES_PER];

    const int g     = blockIdx.x;
    const int tid   = threadIdx.x;
    const int nbase = g * NODES_PER;
    const int ebase = g * EDGES_PER;
    const int NE    = EDGES_PER + NODES_PER;              // edges + self loops

    // Stage h for this graph: 32 KB via async global->LDS, 16B per lane.
    {
        const unsigned lds0 = (unsigned)(size_t)(&h_s[0]);
        const unsigned long long g0 = (unsigned long long)(const void*)(h + (size_t)nbase * H_N);
        for (int i = tid; i < (NODES_PER * H_N) / 4; i += blockDim.x)
            async_copy_b128(lds0 + i * 16u, g0 + (unsigned long long)i * 16ull);
        wait_asynccnt0();
    }
    for (int j = tid; j < NE; j += blockDim.x) {
        if (j < EDGES_PER) {
            src_s[j] = edge_index[ebase + j] - nbase;
            dst_s[j] = edge_index[N_EDGES + ebase + j] - nbase;
        } else {
            src_s[j] = j - EDGES_PER;
            dst_s[j] = j - EDGES_PER;
        }
    }
    __syncthreads();

    if (tid < NODES_PER) {
        float a = 0.f, b = 0.f;
        for (int k = 0; k < H_N; ++k) {
            const float hv = h_s[tid * H_N + k];
            a += hv * Wa[k];
            b += hv * Wa[H_N + k];
        }
        dA[tid] = a; dB[tid] = b;
    }
    __syncthreads();

    const float ba0 = ba[0];
    for (int j = tid; j < NE; j += blockDim.x) {
        float s = dA[src_s[j]] + dB[dst_s[j]] + ba0;
        score_s[j] = s > 0.f ? s : 0.f;
    }
    __syncthreads();

    if (tid < NODES_PER) {                                // self loop => non-empty
        float m = -1e30f;
        for (int j = 0; j < NE; ++j)
            if (src_s[j] == tid && score_s[j] > m) m = score_s[j];
        float d = 0.f;
        for (int j = 0; j < NE; ++j)
            if (src_s[j] == tid) d += __expf(score_s[j] - m);
        mx[tid] = m; den[tid] = d;
    }
    __syncthreads();

    for (int j = tid; j < NE; j += blockDim.x) {
        const int v = src_s[j];
        coeff_s[j] = __expf(score_s[j] - mx[v]) / den[v];
    }
    __syncthreads();

    for (int idx = tid; idx < NODES_PER * H_N; idx += blockDim.x) {
        const int v = idx >> 7;
        const int f = idx & (H_N - 1);
        float acc = 0.f;
        for (int j = 0; j < NE; ++j)
            if (src_s[j] == v) acc += coeff_s[j] * h_s[dst_s[j] * H_N + f];
        nodes_out[(nbase + v) * H_N + f] = acc;
    }
}

// ---------------------------------------------------------------------------
// Line-graph edge attention, one block per graph. The E x E mask is block-
// diagonal (edges stay in-graph) -> 128x128 attention block per graph, and
// left + right = (attL + attR) @ ef: ONE 128x128x128 WMMA GEMM per graph.
// B operand comes from efT (bf16, [feat][edge]) -> contiguous 32B loads.
// ---------------------------------------------------------------------------
__global__ void edge_update(const float*  __restrict__ h,
                            const float*  __restrict__ ef,
                            const __bf16* __restrict__ efT,
                            const int*    __restrict__ edge_index,
                            const float*  __restrict__ Wb,
                            const float*  __restrict__ bb,
                            float* __restrict__ edges_out) {
    __shared__ __bf16 A_s[EDGES_PER * EDGES_PER];         // 32 KB (bf16 att)
    __shared__ int   src_s[EDGES_PER], dst_s[EDGES_PER];
    __shared__ float sL[EDGES_PER], sR[EDGES_PER];
    __shared__ float hdot[NODES_PER];

    const int g     = blockIdx.x;
    const int tid   = threadIdx.x;
    const int nbase = g * NODES_PER;
    const int ebase = g * EDGES_PER;

    for (int j = tid; j < EDGES_PER; j += blockDim.x) {
        src_s[j] = edge_index[ebase + j] - nbase;
        dst_s[j] = edge_index[N_EDGES + ebase + j] - nbase;
    }
    if (tid < NODES_PER) {                                // h[v] . Wb[128:256]
        float a = 0.f;
        for (int k = 0; k < H_N; ++k)
            a += h[(nbase + tid) * H_N + k] * Wb[H_E + k];
        hdot[tid] = a;
    }
    __syncthreads();

    const float bb0 = bb[0];
    for (int j = tid; j < EDGES_PER; j += blockDim.x) {   // ef[j] . Wb[0:128]
        float e = 0.f;
        for (int k = 0; k < H_E; ++k)
            e += ef[(ebase + j) * H_E + k] * Wb[k];
        float l = e + hdot[src_s[j]] + bb0;               // leaky(relu)=relu
        float r = e + hdot[dst_s[j]] + bb0;
        sL[j] = l > 0.f ? l : 0.f;
        sR[j] = r > 0.f ? r : 0.f;
    }
    __syncthreads();

    // Row i of combined attention matrix: A[i][j] = attL + attR (bf16 in LDS)
    for (int i = tid; i < EDGES_PER; i += blockDim.x) {
        const int si = src_s[i], di = dst_s[i];
        float mL = -1e30f, mR = -1e30f;
        bool anyL = false, anyR = false;
        for (int j = 0; j < EDGES_PER; ++j) {
            if (di == src_s[j] && si != dst_s[j]) { anyL = true; if (sL[j] > mL) mL = sL[j]; }
            if (si == dst_s[j] && di != src_s[j]) { anyR = true; if (sR[j] > mR) mR = sR[j]; }
        }
        if (!anyL) mL = 0.f;
        if (!anyR) mR = 0.f;
        float dL = 0.f, dR = 0.f;
        for (int j = 0; j < EDGES_PER; ++j) {
            if (di == src_s[j] && si != dst_s[j]) dL += __expf(sL[j] - mL);
            if (si == dst_s[j] && di != src_s[j]) dR += __expf(sR[j] - mR);
        }
        const float rdL = dL > 0.f ? 1.f / dL : 0.f;      // empty rows -> zeros
        const float rdR = dR > 0.f ? 1.f / dR : 0.f;
        for (int j = 0; j < EDGES_PER; ++j) {
            float a = 0.f;
            if (di == src_s[j] && si != dst_s[j]) a += __expf(sL[j] - mL) * rdL;
            if (si == dst_s[j] && di != src_s[j]) a += __expf(sR[j] - mR) * rdR;
            A_s[i * EDGES_PER + j] = f2bf(a);
        }
    }
    __syncthreads();

    // edges_out[g] = A_s(128x128) @ ef_g(128x128) via WMMA; 8x8 tiles, 8 waves.
    const int lane = tid & 31;
    const int warp = tid >> 5;
    const int half = lane >> 4;
    const int l15  = lane & 15;
    for (int t = warp; t < 64; t += 8) {
        const int m0 = (t >> 3) << 4;
        const int n0 = (t & 7) << 4;
        f32x8 acc = {};
#pragma unroll
        for (int kk = 0; kk < EDGES_PER; kk += 32) {
            const int abase = kk + (half ? 16 : 0);
            bf16x8 alo = *(const bf16x8*)(A_s + (m0 + l15) * EDGES_PER + abase);
            bf16x8 ahi = *(const bf16x8*)(A_s + (m0 + l15) * EDGES_PER + abase + 8);
            bf16x16 a  = __builtin_shufflevector(alo, ahi, 0,1,2,3,4,5,6,7,
                                                 8,9,10,11,12,13,14,15);
            const int bbase = kk + (half ? 16 : 0);
            bf16x16 b = *(const bf16x16*)(efT + (size_t)(n0 + l15) * N_EDGES + ebase + bbase);
            acc = __builtin_amdgcn_wmma_f32_16x16x32_bf16(false, a, false, b,
                                                          (short)0, acc, false, false);
        }
        const int col = n0 + l15;
#pragma unroll
        for (int r = 0; r < 8; ++r) {
            const int row = m0 + r + (half ? 8 : 0);
            edges_out[(ebase + row) * H_E + col] = acc[r];
        }
    }
}

// ---------------------------------------------------------------------------
// Launch. Inputs (setup_inputs order):
//  0:x 1:edge_index 2:edge_attr 3:batch 4:Wh 5:bh 6:Wa 7:ba 8:We 9:be 10:Wb 11:bb
// Output: nodes_out [4096*128] ++ edges_out [8192*128] (f32).
// Workspace: h f32 2MB | ef f32 4MB | efT bf16 2MB | xb .5MB | eab .5MB |
//            WhT 16KB | WeT 8KB  (~9.1 MB total)
// ---------------------------------------------------------------------------
extern "C" void kernel_launch(void* const* d_in, const int* in_sizes, int n_in,
                              void* d_out, int out_size, void* d_ws, size_t ws_size,
                              hipStream_t stream) {
    const float* x  = (const float*)d_in[0];
    const int*   ei = (const int*)  d_in[1];
    const float* ea = (const float*)d_in[2];
    const float* Wh = (const float*)d_in[4];
    const float* bh = (const float*)d_in[5];
    const float* Wa = (const float*)d_in[6];
    const float* ba = (const float*)d_in[7];
    const float* We = (const float*)d_in[8];
    const float* be = (const float*)d_in[9];
    const float* Wb = (const float*)d_in[10];
    const float* bb = (const float*)d_in[11];

    float* nodes_out = (float*)d_out;
    float* edges_out = nodes_out + (size_t)N_NODES * H_N;

    float*  h   = (float*)d_ws;                                  // 4096*128 f32
    float*  ef  = h  + (size_t)N_NODES * H_N;                    // 8192*128 f32
    __bf16* efT = (__bf16*)(ef + (size_t)N_EDGES * H_E);         // [128][8192] bf16
    __bf16* xb  = efT + (size_t)H_E * N_EDGES;                   // 4096*64
    __bf16* eab = xb  + (size_t)N_NODES * F_N;                   // 8192*32
    __bf16* WhT = eab + (size_t)N_EDGES * F_E;                   // [128][64]
    __bf16* WeT = WhT + (size_t)H_N * F_N;                       // [128][32]

    // 0) fp32 -> bf16 conversions + weight transposes (one pass)
    prep_bf16<<<128, 256, 0, stream>>>(x, ea, Wh, We, xb, eab, WhT, WeT);
    // 1) h = relu(x @ Wh + bh): 4096x64x128 -> 2048 tiles, 8 waves/block
    gemm_relu_wmma<F_N><<<256, 256, 0, stream>>>(xb, WhT, bh, h, (__bf16*)nullptr,
                                                 N_NODES, H_N);
    // 2) ef = relu(edge_attr @ We + be): 8192x32x128 -> 4096 tiles (+ efT bf16)
    gemm_relu_wmma<F_E><<<512, 256, 0, stream>>>(eab, WeT, be, ef, efT,
                                                 N_EDGES, H_E);
    // 3) node GAT (one block per graph, async h staging)
    node_update<<<N_GRAPHS, 256, 0, stream>>>(h, ei, Wa, ba, nodes_out);
    // 4) line-graph edge attention + WMMA aggregation (one block per graph)
    edge_update<<<N_GRAPHS, 256, 0, stream>>>(h, ef, efT, ei, Wb, bb, edges_out);

    (void)in_sizes; (void)n_in; (void)out_size; (void)ws_size;
}
